// RefineDet_10204842295989
// MI455X (gfx1250) — matrix operand order repaired and checked by reading, not compiled, not run on hardware
//
#include <hip/hip_runtime.h>
#include <math.h>

typedef __attribute__((ext_vector_type(16))) _Float16 v16h;
typedef __attribute__((ext_vector_type(8)))  float    v8f;

#define BN_INV_F 0.9999950000374997f

// ---------------------------------------------------------------------------
// Implicit-GEMM conv with WMMA f16 -> f32, fused (acc+b0)*s*smul+b1 [+relu]
// Activations: NHWC f16 (Cin a power of two). Weights: packed [Cout][Kpad]
// f16, K order = ((kh*KW+kw)*Cin + cin), Kpad = roundup32(K).
// Templated on square kernel size KS (1 or 3) so kh/kw math is constant.
// ---------------------------------------------------------------------------
struct ConvP {
  const _Float16* x;
  const _Float16* wp;
  const float* b0;
  const float* s;
  const float* b1;
  _Float16* y;
  float* yf;                 // optional f32 NHWC output (nullptr if unused)
  long long yf_bstride;      // element stride between batches in yf
  int Bn, H, W, Cin, cinSh, Cout, OH, OW, stride, pad, dil, K, Kpad, relu;
  float smul;
};

template <int KS>
__device__ __forceinline__ void conv_load_stage(const ConvP& p, _Float16* As,
                                                _Float16* Bs, int n0, int kt,
                                                const int* rB, const int* rOy,
                                                const int* rOx) {
  const int tid = threadIdx.x;
  const int k0 = kt << 5;
  // k-pair handled by this thread is constant: kp = tid & 15
  const int kg = k0 + ((tid & 15) << 1);
  const bool kok = kg < p.K;
  int ci = 0, offY = 0, offX = 0;
  if (kok) {
    int kk = kg >> p.cinSh;           // (kh*KS + kw), Cin is a power of two
    ci = kg & (p.Cin - 1);
    int kh = (KS == 1) ? 0 : (kk / KS);  // constant divisor -> mul/shift
    int kw2 = kk - kh * KS;
    offY = kh * p.dil - p.pad;
    offX = kw2 * p.dil - p.pad;
  }
  unsigned* Au = (unsigned*)As;
#pragma unroll
  for (int j = 0; j < 8; ++j) {
    unsigned val = 0u;
    const int b = rB[j];
    if (b >= 0 && kok) {
      int iy = rOy[j] + offY;
      int ix = rOx[j] + offX;
      if (iy >= 0 && iy < p.H && ix >= 0 && ix < p.W)
        val = *(const unsigned*)(p.x +
              ((((long long)b * p.H + iy) * p.W + ix) * p.Cin + ci));
    }
    Au[tid + (j << 8)] = val;
  }
  // B tile: thread -> row n = tid>>1, half-row (tid&1), 16 halfs contiguous
  {
    int n = n0 + (tid >> 1);
    int cpart = tid & 1;
    uint4 v0 = {0, 0, 0, 0}, v1 = {0, 0, 0, 0};
    if (n < p.Cout) {
      const _Float16* src = p.wp + ((long long)n * p.Kpad + k0 + (cpart << 4));
      v0 = *(const uint4*)src;
      v1 = *(const uint4*)(src + 8);
      __builtin_prefetch(src + 32, 0, 0);   // next k-stage -> global_prefetch_b8
    }
    uint4* dst = (uint4*)(Bs + ((tid >> 1) * 32 + (cpart << 4)));
    dst[0] = v0;
    dst[1] = v1;
  }
}

__device__ __forceinline__ void conv_compute(const _Float16* As,
                                             const _Float16* Bs, int wm, int wn,
                                             int lane, v8f acc[2][4]) {
  const int mr = lane & 15;
  const int hf = lane >> 4;
  union VH { v16h v; uint4 q[2]; };
  VH a[2], b[4];
#pragma unroll
  for (int i = 0; i < 2; ++i) {
    const _Float16* ap = As + (wm * 32 + i * 16 + mr) * 32;
    a[i].q[0] = *(const uint4*)(ap + hf * 8);        // K = 8*hf .. +7
    a[i].q[1] = *(const uint4*)(ap + 16 + hf * 8);   // K = 16+8*hf .. +7
  }
#pragma unroll
  for (int j = 0; j < 4; ++j) {
    const _Float16* bp = Bs + (wn * 64 + j * 16 + mr) * 32;
    b[j].q[0] = *(const uint4*)(bp + hf * 8);        // K = 8*hf .. +7
    b[j].q[1] = *(const uint4*)(bp + 16 + hf * 8);   // K = 16+8*hf .. +7
  }
#pragma unroll
  for (int i = 0; i < 2; ++i)
#pragma unroll
    for (int j = 0; j < 4; ++j)
      acc[i][j] = __builtin_amdgcn_wmma_f32_16x16x32_f16(
          false, a[i].v, false, b[j].v, (short)0, acc[i][j], false, false);
}

template <int KS>
__global__ __launch_bounds__(256) void k_conv_wmma(ConvP p) {
  __shared__ _Float16 As[2][4096];
  __shared__ _Float16 Bs[2][4096];
  const int m0 = blockIdx.x << 7;
  const int n0 = blockIdx.y << 7;
  const int lane = threadIdx.x & 31;
  const int wv = threadIdx.x >> 5;
  const int wm = wv & 3, wn = wv >> 2;
  const int OHW = p.OH * p.OW;
  const int M = p.Bn * OHW;

  // Hoisted, k-invariant row decomposition for this thread's 8 A-tile rows.
  int rB[8], rOy[8], rOx[8];
#pragma unroll
  for (int j = 0; j < 8; ++j) {
    int m = m0 + (threadIdx.x >> 4) + j * 16;
    if (m < M) {
      int b = m / OHW;
      int r = m - b * OHW;
      int oy = r / p.OW;
      int ox = r - oy * p.OW;
      rB[j] = b;
      rOy[j] = oy * p.stride;
      rOx[j] = ox * p.stride;
    } else {
      rB[j] = -1; rOy[j] = 0; rOx[j] = 0;
    }
  }

  v8f acc[2][4] = {};
  const int nkt = (p.K + 31) >> 5;
  conv_load_stage<KS>(p, As[0], Bs[0], n0, 0, rB, rOy, rOx);
  __syncthreads();
  for (int kt = 0; kt < nkt; ++kt) {
    const int cb = kt & 1;
    if (kt + 1 < nkt)
      conv_load_stage<KS>(p, As[cb ^ 1], Bs[cb ^ 1], n0, kt + 1, rB, rOy, rOx);
    conv_compute(As[cb], Bs[cb], wm, wn, lane, acc);
    __syncthreads();
  }

  const int mr = lane & 15;
  const int hf = lane >> 4;
#pragma unroll
  for (int i = 0; i < 2; ++i) {
#pragma unroll
    for (int j = 0; j < 4; ++j) {
      int n = n0 + wn * 64 + j * 16 + mr;
      if (n >= p.Cout) continue;
      float b0v = p.b0 ? p.b0[n] : 0.f;
      float sv = p.s ? p.s[n] * p.smul : 1.f;
      float b1v = p.b1 ? p.b1[n] : 0.f;
#pragma unroll
      for (int e = 0; e < 8; ++e) {
        int m = m0 + wm * 32 + i * 16 + hf * 8 + e;
        if (m >= M) continue;
        float v = (acc[i][j][e] + b0v) * sv + b1v;
        if (p.relu) v = fmaxf(v, 0.f);
        int b = m / OHW;
        int r = m - b * OHW;
        p.y[((long long)b * OHW + r) * p.Cout + n] = (_Float16)v;
        if (p.yf)
          p.yf[(long long)b * p.yf_bstride + (long long)r * p.Cout + n] = v;
      }
    }
  }
}

// ---------------------------------------------------------------------------
// Packing / elementwise / pooling / norm kernels
// ---------------------------------------------------------------------------
__global__ void k_pack_input(const float* x, _Float16* y, int Bn, int HW) {
  long long i = (long long)blockIdx.x * blockDim.x + threadIdx.x;
  long long tot = (long long)Bn * HW * 4;
  if (i >= tot) return;
  int c = (int)(i & 3);
  long long p = i >> 2;
  int b = (int)(p / HW);
  int r = (int)(p - (long long)b * HW);
  float v = (c < 3) ? x[((long long)b * 3 + c) * HW + r] : 0.f;
  y[i] = (_Float16)v;
}

__global__ void k_pack_weight(const float* w, _Float16* wp, int Cout, int Cin,
                              int CinP, int KH, int KW, int Kpad) {
  long long i = (long long)blockIdx.x * blockDim.x + threadIdx.x;
  if (i >= (long long)Cout * Kpad) return;
  int co = (int)(i / Kpad);
  int k = (int)(i - (long long)co * Kpad);
  float v = 0.f;
  if (k < KH * KW * CinP) {
    int kk = k / CinP;
    int c = k - kk * CinP;
    int kh = kk / KW;
    int kw2 = kk - kh * KW;
    if (c < Cin) v = w[(((long long)co * Cin + c) * KH + kh) * KW + kw2];
  }
  wp[i] = (_Float16)v;
}

__global__ void k_maxpool2(const _Float16* x, _Float16* y, int Bn, int H, int W,
                           int C) {
  int OH = H >> 1, OW = W >> 1;
  long long i = (long long)blockIdx.x * blockDim.x + threadIdx.x;
  long long tot = (long long)Bn * OH * OW * C;
  if (i >= tot) return;
  int c = (int)(i % C);
  long long p = i / C;
  int ox = (int)(p % OW); p /= OW;
  int oy = (int)(p % OH);
  int b = (int)(p / OH);
  long long base = (((long long)b * H + 2 * oy) * W + 2 * ox) * C + c;
  float m0 = (float)x[base], m1 = (float)x[base + C];
  float m2 = (float)x[base + (long long)W * C], m3 = (float)x[base + (long long)W * C + C];
  y[i] = (_Float16)fmaxf(fmaxf(m0, m1), fmaxf(m2, m3));
}

__global__ void k_l2norm(const _Float16* x, const float* w, _Float16* y, int C) {
  __shared__ float red[256];
  long long pix = blockIdx.x;
  const _Float16* xp = x + pix * C;
  float s = 0.f;
  for (int c = threadIdx.x; c < C; c += 256) {
    float v = (float)xp[c];
    s += v * v;
  }
  red[threadIdx.x] = s;
  __syncthreads();
  for (int st = 128; st > 0; st >>= 1) {
    if ((int)threadIdx.x < st) red[threadIdx.x] += red[threadIdx.x + st];
    __syncthreads();
  }
  float inv = 1.f / (sqrtf(red[0]) + 1e-10f);
  for (int c = threadIdx.x; c < C; c += 256)
    y[pix * C + c] = (_Float16)((float)xp[c] * inv * w[c]);
}

__global__ void k_relu(const _Float16* x, _Float16* y, long long n) {
  long long i = (long long)blockIdx.x * blockDim.x + threadIdx.x;
  if (i >= n) return;
  y[i] = (_Float16)fmaxf((float)x[i], 0.f);
}

// y = relu(s + deconv2(xin)) ; deconv: 2x2 stride-2 transposed conv (pad 0)
__global__ void k_deconv_add_relu(const _Float16* s, const _Float16* xin,
                                  const float* w, const float* bias,
                                  _Float16* y, int Bn, int OH, int OW, int C) {
  long long i = (long long)blockIdx.x * blockDim.x + threadIdx.x;
  long long tot = (long long)Bn * OH * OW * C;
  if (i >= tot) return;
  int co = (int)(i % C);
  long long p = i / C;
  int ox = (int)(p % OW); p /= OW;
  int oy = (int)(p % OH);
  int b = (int)(p / OH);
  int IH = OH >> 1, IW = OW >> 1;
  int kh = (oy + 1) & 1, kw2 = (ox + 1) & 1;
  int iy = (oy - 1 + kh) >> 1, ix = (ox - 1 + kw2) >> 1;
  float acc = bias[co];
  if (iy >= 0 && iy < IH && ix >= 0 && ix < IW) {
    const _Float16* xp = xin + (((long long)b * IH + iy) * IW + ix) * C;
    const float* wp = w + ((long long)(1 - kh) * 2 + (1 - kw2));
    for (int ci = 0; ci < C; ++ci)
      acc += (float)xp[ci] * wp[((long long)ci * C + co) * 4];
  }
  y[i] = (_Float16)fmaxf((float)s[i] + acc, 0.f);
}

__global__ void k_gn_relu(const _Float16* x, const float* g, const float* be,
                          _Float16* y, int HW, int C, int cg, float eps) {
  __shared__ float r1[256], r2[256];
  int groups = C / cg;
  int bg = blockIdx.x;
  int b = bg / groups;
  int grp = bg - b * groups;
  long long base = (long long)b * HW * C + grp * cg;
  int n = HW * cg;
  float s = 0.f, ss = 0.f;
  for (int i = threadIdx.x; i < n; i += 256) {
    int pix = i / cg;
    int cc = i - pix * cg;
    float v = (float)x[base + (long long)pix * C + cc];
    s += v;
    ss += v * v;
  }
  r1[threadIdx.x] = s;
  r2[threadIdx.x] = ss;
  __syncthreads();
  for (int st = 128; st > 0; st >>= 1) {
    if ((int)threadIdx.x < st) {
      r1[threadIdx.x] += r1[threadIdx.x + st];
      r2[threadIdx.x] += r2[threadIdx.x + st];
    }
    __syncthreads();
  }
  float mean = r1[0] / n;
  float var = r2[0] / n - mean * mean;
  float rs = rsqrtf(var + eps);
  for (int i = threadIdx.x; i < n; i += 256) {
    int pix = i / cg;
    int cc = i - pix * cg;
    int c = grp * cg + cc;
    long long idx = base + (long long)pix * C + cc;
    float v = ((float)x[idx] - mean) * rs * g[c] + be[c];
    y[idx] = (_Float16)fmaxf(v, 0.f);
  }
}

__global__ void k_att(const _Float16* s, const _Float16* att, _Float16* y,
                      long long npix, int C, int A) {
  long long pix = (long long)blockIdx.x * blockDim.x + threadIdx.x;
  if (pix >= npix) return;
  const _Float16* ap = att + pix * A;
  float m = 1.f / (1.f + expf(-(float)ap[0]));
  for (int i = 1; i < A; ++i) {
    float v = 1.f / (1.f + expf(-(float)ap[i]));
    m = fmaxf(m, v);
  }
  const _Float16* sp = s + pix * C;
  _Float16* yp = y + pix * C;
  for (int c = 0; c < C; ++c) yp[c] = (_Float16)((float)sp[c] * m);
}

__global__ void k_make_offsets(const float* loc, long long bstride, float* offs,
                               int Bn, int H, int W) {
  long long i = (long long)blockIdx.x * blockDim.x + threadIdx.x;
  long long HW = (long long)H * W;
  if (i >= (long long)Bn * HW) return;
  int b = (int)(i / HW);
  int r = (int)(i - (long long)b * HW);
  int yy = r / W;
  int xx = r - yy * W;
  const float* lp = loc + (long long)b * bstride + (long long)r * 12;
  const float AW[3] = {4.f, 5.65685424949238f, 2.82842712474619f};
  const float AH[3] = {4.f, 2.82842712474619f, 5.65685424949238f};
  for (int a = 0; a < 3; ++a) {
    float d0 = lp[a * 4 + 0], d1 = lp[a * 4 + 1];
    float d2 = lp[a * 4 + 2], d3 = lp[a * 4 + 3];
    float cx = xx + 0.5f + d0 * 0.1f * AW[a];
    float cy = yy + 0.5f + d1 * 0.1f * AH[a];
    float ww = AW[a] * expf(d2 * 0.2f);
    float hh = AH[a] * expf(d3 * 0.2f);
    float* op = offs + (((long long)a * Bn + b) * HW + r) * 18;
    for (int k = 0; k < 9; ++k) {
      float by = (float)(k / 3) - 1.f, bx = (float)(k % 3) - 1.f;
      float py = cy + by * hh * (1.f / 3.f);
      float px = cx + bx * ww * (1.f / 3.f);
      op[2 * k] = py - ((float)yy + by);
      op[2 * k + 1] = px - ((float)xx + bx);
    }
  }
}

// One block per (b,pixel) for a single anchor: bilinear sample 9x256 into LDS,
// then compute 4 loc + 21 conf outputs cooperatively.
__global__ __launch_bounds__(256) void k_deform(
    const _Float16* feat, const float* off, const float* wl, const float* bl,
    const float* wc, const float* bc, float* outloc, int locC, int locOff,
    float* outconf, int confC, int confOff, long long bstride, int H, int W) {
  __shared__ float samp[9 * 256];
  __shared__ float red[256];
  long long HW = (long long)H * W;
  int b = (int)(blockIdx.x / HW);
  int r = (int)(blockIdx.x - (long long)b * HW);
  int yy = r / W;
  int xx = r - yy * W;
  int c = threadIdx.x;
  const float* op = off + ((long long)b * HW + r) * 18;
  for (int k = 0; k < 9; ++k) {
    float by = (float)(k / 3) - 1.f, bx = (float)(k % 3) - 1.f;
    float py = (float)yy + by + op[2 * k];
    float px = (float)xx + bx + op[2 * k + 1];
    float fy = floorf(py), fx = floorf(px);
    int y0 = (int)fy, x0 = (int)fx;
    float wy = py - fy, wx = px - fx;
    auto tap = [&](int yi, int xi) -> float {
      if (yi < 0 || yi >= H || xi < 0 || xi >= W) return 0.f;
      return (float)feat[(((long long)b * H + yi) * W + xi) * 256 + c];
    };
    samp[k * 256 + c] = tap(y0, x0) * (1.f - wy) * (1.f - wx) +
                        tap(y0, x0 + 1) * (1.f - wy) * wx +
                        tap(y0 + 1, x0) * wy * (1.f - wx) +
                        tap(y0 + 1, x0 + 1) * wy * wx;
  }
  __syncthreads();
  int o = threadIdx.x >> 3;
  int part = threadIdx.x & 7;
  float partial = 0.f;
  if (o < 25) {
    const float* wsel = (o < 4) ? (wl + (long long)o * 2304)
                                : (wc + (long long)(o - 4) * 2304);
    for (int q = part * 288; q < part * 288 + 288; ++q) {
      int k = q >> 8;
      int cc = q & 255;
      partial += samp[q] * wsel[cc * 9 + k];
    }
  }
  red[threadIdx.x] = partial;
  __syncthreads();
  if (o < 25 && part == 0) {
    float s = 0.f;
    for (int t = 0; t < 8; ++t) s += red[o * 8 + t];
    if (o < 4)
      outloc[(long long)b * bstride + (long long)r * locC + locOff + o] = s + bl[o];
    else
      outconf[(long long)b * bstride + (long long)r * confC + confOff + (o - 4)] =
          s + bc[o - 4];
  }
}

// ---------------------------------------------------------------------------
// Host orchestration
// ---------------------------------------------------------------------------
struct Fm {
  _Float16* p;
  int H, W, C;
};

extern "C" void kernel_launch(void* const* d_in, const int* in_sizes, int n_in,
                              void* d_out, int out_size, void* d_ws,
                              size_t ws_size, hipStream_t stream) {
  (void)in_sizes; (void)n_in; (void)out_size; (void)ws_size;
  const int Bn = 4;
  const float* X = (const float*)d_in[0];
  auto IN = [&](int i) { return (const float*)d_in[i]; };

  size_t cur = 0;
  auto alloc = [&](size_t bytes) -> void* {
    cur = (cur + 255) & ~(size_t)255;
    void* r = (char*)d_ws + cur;
    cur += bytes;
    return r;
  };

  const size_t bigBytes = (size_t)Bn * 320 * 320 * 64 * sizeof(_Float16);
  _Float16* ping = (_Float16*)alloc(bigBytes);
  _Float16* pong = (_Float16*)alloc(bigBytes);
  _Float16* src0 = (_Float16*)alloc((size_t)Bn * 40 * 40 * 512 * 2);
  _Float16* src1 = (_Float16*)alloc((size_t)Bn * 20 * 20 * 512 * 2);
  _Float16* src2 = (_Float16*)alloc((size_t)Bn * 10 * 10 * 1024 * 2);
  _Float16* src3 = (_Float16*)alloc((size_t)Bn * 5 * 5 * 512 * 2);

  auto conv = [&](Fm in, _Float16* out, int Cout, int k, int stride, int pad,
                  int dil, const float* wraw, int cinOrig, const float* b0,
                  const float* s, const float* b1, float smul, int relu,
                  float* yf, long long yfstride) -> Fm {
    int Cin = in.C;
    int K = k * k * Cin;
    int Kpad = (K + 31) & ~31;
    _Float16* wp = (_Float16*)alloc((size_t)Cout * Kpad * sizeof(_Float16));
    long long wtot = (long long)Cout * Kpad;
    k_pack_weight<<<dim3((unsigned)((wtot + 255) / 256)), 256, 0, stream>>>(
        wraw, wp, Cout, cinOrig, Cin, k, k, Kpad);
    int OH = (in.H + 2 * pad - dil * (k - 1) - 1) / stride + 1;
    int OW = (in.W + 2 * pad - dil * (k - 1) - 1) / stride + 1;
    ConvP cp;
    cp.x = in.p; cp.wp = wp; cp.b0 = b0; cp.s = s; cp.b1 = b1;
    cp.y = out; cp.yf = yf; cp.yf_bstride = yfstride;
    cp.Bn = Bn; cp.H = in.H; cp.W = in.W; cp.Cin = Cin;
    cp.cinSh = __builtin_ctz((unsigned)Cin);
    cp.Cout = Cout; cp.OH = OH; cp.OW = OW;
    cp.stride = stride; cp.pad = pad; cp.dil = dil;
    cp.K = K; cp.Kpad = Kpad; cp.relu = relu; cp.smul = smul;
    dim3 g((unsigned)((Bn * OH * OW + 127) / 128), (unsigned)((Cout + 127) / 128));
    if (k == 1)
      k_conv_wmma<1><<<g, 256, 0, stream>>>(cp);
    else
      k_conv_wmma<3><<<g, 256, 0, stream>>>(cp);
    return Fm{out, OH, OW, Cout};
  };

  auto pool = [&](Fm in, _Float16* out) -> Fm {
    int OH = in.H >> 1, OW = in.W >> 1;
    long long tot = (long long)Bn * OH * OW * in.C;
    k_maxpool2<<<(unsigned)((tot + 255) / 256), 256, 0, stream>>>(
        in.p, out, Bn, in.H, in.W, in.C);
    return Fm{out, OH, OW, in.C};
  };

  int vi = 0;
  auto vggconv = [&](Fm in, _Float16* out, int Cout, int k, int pad, int dil) -> Fm {
    int base = 1 + vi * 4;
    int cinOrig = (vi == 0) ? 3 : in.C;
    vi++;
    return conv(in, out, Cout, k, 1, pad, dil, IN(base), cinOrig, IN(base + 1),
                IN(base + 2), IN(base + 3), BN_INV_F, 1, nullptr, 0);
  };

  // ---- input pack (NCHW f32 -> NHWC f16, C 3->4) ----
  const int HW320 = 320 * 320;
  long long tot0 = (long long)Bn * HW320 * 4;
  k_pack_input<<<(unsigned)((tot0 + 255) / 256), 256, 0, stream>>>(X, ping, Bn, HW320);
  Fm a{ping, 320, 320, 4};

  // ---- VGG backbone ----
  a = vggconv(a, pong, 64, 3, 1, 1);
  a = vggconv(a, ping, 64, 3, 1, 1);
  a = pool(a, pong);                       // 160
  a = vggconv(a, ping, 128, 3, 1, 1);
  a = vggconv(a, pong, 128, 3, 1, 1);
  a = pool(a, ping);                       // 80
  a = vggconv(a, pong, 256, 3, 1, 1);
  a = vggconv(a, ping, 256, 3, 1, 1);
  a = vggconv(a, pong, 256, 3, 1, 1);
  a = pool(a, ping);                       // 40
  a = vggconv(a, pong, 512, 3, 1, 1);
  a = vggconv(a, ping, 512, 3, 1, 1);
  a = vggconv(a, pong, 512, 3, 1, 1);      // conv4_3
  k_l2norm<<<(unsigned)(Bn * 40 * 40), 256, 0, stream>>>(a.p, IN(177), src0, 512);
  a = pool(a, ping);                       // 20
  a = vggconv(a, pong, 512, 3, 1, 1);
  a = vggconv(a, ping, 512, 3, 1, 1);
  a = vggconv(a, pong, 512, 3, 1, 1);      // conv5_3
  k_l2norm<<<(unsigned)(Bn * 20 * 20), 256, 0, stream>>>(a.p, IN(178), src1, 512);
  a = pool(a, ping);                       // 10
  a = vggconv(a, pong, 1024, 3, 6, 6);     // conv6, dilated
  a = vggconv(a, src2, 1024, 1, 0, 1);     // conv7 1x1
  Fm src2fm = a;
  a = conv(a, ping, 256, 1, 1, 0, 1, IN(61), a.C, IN(62), IN(63), IN(64),
           BN_INV_F, 1, nullptr, 0);
  a = conv(a, src3, 512, 3, 2, 1, 1, IN(65), a.C, IN(66), IN(67), IN(68),
           BN_INV_F, 1, nullptr, 0);       // 5x5
  Fm srcs[4] = {Fm{src0, 40, 40, 512}, Fm{src1, 20, 20, 512}, src2fm, a};

  // ---- output layout ----
  float* dof = (float*)d_out;
  const long long TOT = 197625;
  long long Pl[4] = {1600, 400, 100, 25};
  long long aL[4], aC[4], dL[4], dC[4];
  {
    long long o1 = 0, o2 = 0, o3 = 0, o4 = 0;
    for (int l = 0; l < 4; ++l) {
      aL[l] = o1; o1 += Pl[l] * 12;
      aC[l] = o2; o2 += Pl[l] * 6;
      dL[l] = o3; o3 += Pl[l] * 12;
      dC[l] = o4; o4 += Pl[l] * 63;
    }
  }
  const long long armConfBase = 25500, admLocBase = 38250, admConfBase = 63750;

  // ---- ARM heads + anchor offsets ----
  float* offs[4];
  for (int l = 0; l < 4; ++l) {
    Fm s = srcs[l];
    conv(s, pong, 12, 3, 1, 1, 1, IN(69 + 2 * l), s.C, nullptr, nullptr,
         IN(70 + 2 * l), 1.f, 0, dof + aL[l], TOT);
    offs[l] = (float*)alloc((size_t)3 * Bn * Pl[l] * 18 * sizeof(float));
    long long np = (long long)Bn * Pl[l];
    k_make_offsets<<<(unsigned)((np + 255) / 256), 256, 0, stream>>>(
        dof + aL[l], TOT, offs[l], Bn, s.H, s.W);
    conv(s, pong, 6, 3, 1, 1, 1, IN(77 + 2 * l), s.C, nullptr, nullptr,
         IN(78 + 2 * l), 1.f, 0, dof + armConfBase + aC[l], TOT);
  }

  // ---- TCB + cate attention ----
  _Float16* tcbbuf[4];
  _Float16* attsrc[4];
  for (int l = 0; l < 4; ++l) {
    tcbbuf[l] = (_Float16*)alloc((size_t)Bn * Pl[l] * 256 * 2);
    attsrc[l] = (_Float16*)alloc((size_t)Bn * Pl[l] * 256 * 2);
  }
  _Float16* attbuf = (_Float16*)alloc((size_t)Bn * 1600 * 20 * sizeof(_Float16));

  Fm prev{nullptr, 0, 0, 0};
  for (int k2 = 0; k2 < 4; ++k2) {
    int lvl = 3 - k2;
    Fm s = srcs[lvl];
    int bA = 85 + (lvl * 2) * 2;
    int bB = 85 + (lvl * 2 + 1) * 2;
    Fm t = conv(s, ping, 256, 3, 1, 1, 1, IN(bA), s.C, nullptr, nullptr,
                IN(bA + 1), 1.f, 1, nullptr, 0);
    t = conv(t, pong, 256, 3, 1, 1, 1, IN(bB), 256, nullptr, nullptr,
             IN(bB + 1), 1.f, 0, nullptr, 0);
    long long tn = (long long)Bn * t.H * t.W * 256;
    if (k2 == 0) {
      k_relu<<<(unsigned)((tn + 255) / 256), 256, 0, stream>>>(t.p, ping, tn);
    } else {
      int ui = 101 + lvl * 2;
      k_deconv_add_relu<<<(unsigned)((tn + 255) / 256), 256, 0, stream>>>(
          t.p, prev.p, IN(ui), IN(ui + 1), ping, Bn, t.H, t.W, 256);
    }
    Fm pf{ping, t.H, t.W, 256};
    int b2 = 107 + lvl * 2;
    Fm o = conv(pf, tcbbuf[lvl], 256, 3, 1, 1, 1, IN(b2), 256, nullptr, nullptr,
                IN(b2 + 1), 1.f, 1, nullptr, 0);
    prev = o;

    // cate chain (shared 4 convs) + solo attention
    Fm f{tcbbuf[lvl], o.H, o.W, 256};
    for (int ci = 0; ci < 4; ++ci) {
      _Float16* ob = (ci & 1) ? pong : ping;
      f = conv(f, ob, 256, 3, 1, 1, 1, IN(115 + 3 * ci), 256, nullptr, nullptr,
               nullptr, 1.f, 0, nullptr, 0);
      k_gn_relu<<<(unsigned)(Bn * 32), 256, 0, stream>>>(
          f.p, IN(116 + 3 * ci), IN(117 + 3 * ci), f.p, f.H * f.W, 256, 8, 1e-5f);
    }
    conv(f, attbuf, 20, 3, 1, 1, 1, IN(127), 256, nullptr, nullptr, IN(128),
         1.f, 0, nullptr, 0);
    long long np = (long long)Bn * o.H * o.W;
    k_att<<<(unsigned)((np + 255) / 256), 256, 0, stream>>>(
        tcbbuf[lvl], attbuf, attsrc[lvl], np, 256, 20);
  }

  // ---- deformable ADM heads ----
  for (int l = 0; l < 4; ++l) {
    Fm s = srcs[l];
    for (int an = 0; an < 3; ++an) {
      int li = 129 + (an * 4 + l) * 2;
      int ci = 153 + (an * 4 + l) * 2;
      const float* offp = offs[l] + (size_t)an * Bn * Pl[l] * 18;
      k_deform<<<(unsigned)(Bn * Pl[l]), 256, 0, stream>>>(
          attsrc[l], offp, IN(li), IN(li + 1), IN(ci), IN(ci + 1),
          dof + admLocBase + dL[l], 12, an * 4,
          dof + admConfBase + dC[l], 63, an * 21, TOT, s.H, s.W);
    }
  }
}
